// GNN27_27410481283396
// MI455X (gfx1250) — matrix operand order repaired and checked by reading, not compiled, not run on hardware
//
#include <hip/hip_runtime.h>
#include <hip/hip_bf16.h>
#include <math.h>

typedef __attribute__((ext_vector_type(16))) __bf16 v16bf;
typedef __attribute__((ext_vector_type(8)))  float  v8f;
typedef __attribute__((ext_vector_type(4)))  float  v4f;

#define NN 8192

// ---------------------------------------------------------------------------
// pack: Y = X @ W   (X:[8192,Fin] f32, W:[Fin,Fout] f32) -> bf16 in WMMA
// B-operand layout, per 16-column group g:
//   B tile for k-block kb (32 K rows):  lane = n + ((kr>>3)&1 ? 16 : 0), elem = (kr&7) + (kr>=16 ? 8 : 0)
//   flat bf16 index = g*8192*16 + kb*512 + lane*16 + elem
// ---------------------------------------------------------------------------
__global__ void pack_kernel(const float* __restrict__ X, const float* __restrict__ W,
                            __bf16* __restrict__ YB, int Fin, int Fout) {
    int idx = blockIdx.x * blockDim.x + threadIdx.x;
    if (idx >= NN * Fout) return;
    int r = idx / Fout, c = idx % Fout;
    float s = 0.f;
    for (int k = 0; k < Fin; ++k) s += X[(size_t)r * Fin + k] * W[k * Fout + c];
    int g = c >> 4, n2 = c & 15;
    int kb = r >> 5, kr = r & 31;
    int lane = n2 + (((kr >> 3) & 1) ? 16 : 0);
    int e = (kr & 7) + ((kr >= 16) ? 8 : 0);
    YB[(size_t)g * NN * 16 + (size_t)kb * 512 + lane * 16 + e] = (__bf16)s;
}

// ---------------------------------------------------------------------------
// gemm_agg<NG>: H = relu(A @ Y + bias), A:[8192,8192] f32 row-major,
// Y packed bf16 (NG groups of 16 cols), H:[8192, 16*NG] f32 row-major.
// Block = 16 output rows; 8 waves split K; WMMA bf16 16x16x32, fp32 accum.
// A 16x32 bf16 operand layout: lane holds row (lane&15);
//   K(e) = (e<8 ? e : 16+(e&7)) + (lane>=16 ? 8 : 0)
// ---------------------------------------------------------------------------
template <int NG>
__global__ __launch_bounds__(256) void gemm_agg(const float* __restrict__ A,
                                                const __bf16* __restrict__ YB,
                                                const float* __restrict__ bias,
                                                float* __restrict__ H) {
    constexpr int NF = 16 * NG;
    const int lane = threadIdx.x & 31;
    const int wave = threadIdx.x >> 5;
    const int m0   = blockIdx.x * 16;
    const float* arow = A + (size_t)(m0 + (lane & 15)) * NN;
    const int asel = (lane >> 4) * 8;            // lanes 16-31 take K offset +8
    v8f acc[NG];
#pragma unroll
    for (int g = 0; g < NG; ++g) acc[g] = (v8f)(0.f);

    const int kt0 = wave * 32;                    // 256 k-tiles / 8 waves
#pragma unroll 2
    for (int kt = kt0; kt < kt0 + 32; ++kt) {
        const float* ap = arow + kt * 32 + asel;
        v4f a0 = *(const v4f*)(ap);
        v4f a1 = *(const v4f*)(ap + 4);
        v4f a2 = *(const v4f*)(ap + 16);
        v4f a3 = *(const v4f*)(ap + 20);
        v16bf av;
#pragma unroll
        for (int i = 0; i < 4; ++i) {
            av[i]      = (__bf16)a0[i];
            av[4 + i]  = (__bf16)a1[i];
            av[8 + i]  = (__bf16)a2[i];
            av[12 + i] = (__bf16)a3[i];
        }
        const __bf16* bp = YB + (size_t)kt * 512 + lane * 16;
#pragma unroll
        for (int g = 0; g < NG; ++g) {
            v16bf bv = *(const v16bf*)(bp + (size_t)g * NN * 16);
            acc[g] = __builtin_amdgcn_wmma_f32_16x16x32_bf16(
                false, av, false, bv, (short)0, acc[g], false, false);
        }
    }

    // cross-wave K reduction in LDS, then bias + relu + store
    __shared__ float red[8 * 32 * 16];
    float* slot = &red[(wave * 32 + lane) * 16];
#pragma unroll
    for (int g = 0; g < NG; ++g)
#pragma unroll
        for (int r = 0; r < 8; ++r) slot[g * 8 + r] = acc[g][r];
    __syncthreads();

    for (int idx = threadIdx.x; idx < 32 * NG * 8; idx += 256) {
        int l = idx / (NG * 8);
        int rem = idx % (NG * 8);
        int g = rem >> 3, r = rem & 7;
        float s = 0.f;
#pragma unroll
        for (int w = 0; w < 8; ++w) s += red[(w * 32 + l) * 16 + g * 8 + r];
        int mrow = r + ((l >= 16) ? 8 : 0);       // C/D layout: VGPR r -> M
        int col  = g * 16 + (l & 15);
        s += bias[col];
        s = s > 0.f ? s : 0.f;
        H[(size_t)(m0 + mrow) * NF + col] = s;
    }
}

// ---------------------------------------------------------------------------
// escore: E[n,h] = exp(tanh(h_n . W_att[:,h]))  (tanh bounded -> no max sub)
// ---------------------------------------------------------------------------
__global__ __launch_bounds__(256) void escore_kernel(const float* __restrict__ H2i,
                                                     const float* __restrict__ H2n,
                                                     const float* __restrict__ W_att,
                                                     float* __restrict__ E) {
    __shared__ float watt[192];
    if (threadIdx.x < 192) watt[threadIdx.x] = W_att[threadIdx.x];
    __syncthreads();
    int n = blockIdx.x * 256 + threadIdx.x;
    const float* hi = H2i + (size_t)n * 32;
    const float* hn = H2n + (size_t)n * 32;
    float s0 = 0.f, s1 = 0.f, s2 = 0.f;
#pragma unroll 4
    for (int d = 0; d < 64; ++d) {
        float v = (d < 32) ? hi[d] : hn[d - 32];
        s0 += v * watt[d * 3 + 0];
        s1 += v * watt[d * 3 + 1];
        s2 += v * watt[d * 3 + 2];
    }
    E[n * 3 + 0] = expf(tanhf(s0));
    E[n * 3 + 1] = expf(tanhf(s1));
    E[n * 3 + 2] = expf(tanhf(s2));
}

// ---------------------------------------------------------------------------
// pool_partial: per-block partials of wsum[h][d] = sum_n E[n,h]*h[n,d]
// (j<192) and sumexp[h] (j in 192..194). 32 blocks x 256 rows.
// ---------------------------------------------------------------------------
__global__ __launch_bounds__(256) void pool_partial(const float* __restrict__ E,
                                                    const float* __restrict__ H2i,
                                                    const float* __restrict__ H2n,
                                                    float* __restrict__ part) {
    int j = threadIdx.x;
    int r0 = blockIdx.x * 256;
    float acc = 0.f;
    if (j < 192) {
        int h = j / 64, d = j % 64;
        const float* Hs = (d < 32) ? H2i : H2n;
        int dd = d & 31;
        for (int n = r0; n < r0 + 256; ++n)
            acc += E[n * 3 + h] * Hs[(size_t)n * 32 + dd];
    } else if (j < 195) {
        int h = j - 192;
        for (int n = r0; n < r0 + 256; ++n) acc += E[n * 3 + h];
    }
    if (j < 195) part[blockIdx.x * 195 + j] = acc;
}

// ---------------------------------------------------------------------------
// final: reduce 32 partials, pooled = wsum/sumexp, out = pooled . W_dense + b
// ---------------------------------------------------------------------------
__global__ __launch_bounds__(256) void final_kernel(const float* __restrict__ part,
                                                    const float* __restrict__ Wd,
                                                    const float* __restrict__ bd,
                                                    float* __restrict__ out) {
    __shared__ float tot[195];
    __shared__ float prod[192];
    int j = threadIdx.x;
    if (j < 195) {
        float s = 0.f;
        for (int b = 0; b < 32; ++b) s += part[b * 195 + j];
        tot[j] = s;
    }
    __syncthreads();
    if (j < 192) prod[j] = (tot[j] / tot[192 + j / 64]) * Wd[j];
    __syncthreads();
    if (j == 0) {
        float s = bd[0];
        for (int i = 0; i < 192; ++i) s += prod[i];
        out[0] = s;
    }
}

// ---------------------------------------------------------------------------
extern "C" void kernel_launch(void* const* d_in, const int* in_sizes, int n_in,
                              void* d_out, int out_size, void* d_ws, size_t ws_size,
                              hipStream_t stream) {
    (void)in_sizes; (void)n_in; (void)out_size; (void)ws_size;
    const float* x_int   = (const float*)d_in[0];
    const float* x_nh    = (const float*)d_in[1];
    const float* adj_int = (const float*)d_in[2];
    const float* adj_nh  = (const float*)d_in[3];
    const float* W1_int  = (const float*)d_in[4];
    const float* b1_int  = (const float*)d_in[5];
    const float* W1_nh   = (const float*)d_in[6];
    const float* b1_nh   = (const float*)d_in[7];
    const float* W2_int  = (const float*)d_in[8];
    const float* b2_int  = (const float*)d_in[9];
    const float* W2_nh   = (const float*)d_in[10];
    const float* b2_nh   = (const float*)d_in[11];
    const float* W_att   = (const float*)d_in[12];
    const float* W_dense = (const float*)d_in[13];
    const float* b_dense = (const float*)d_in[14];
    float* out = (float*)d_out;

    char* w = (char*)d_ws;
    __bf16* Y1i = (__bf16*)(w + 0);               // 8192*16*2 = 256 KB
    __bf16* Y1n = (__bf16*)(w + 262144);          // 256 KB
    __bf16* Y2i = (__bf16*)(w + 524288);          // 8192*32*2 = 512 KB
    __bf16* Y2n = (__bf16*)(w + 1048576);         // 512 KB
    float*  H1i = (float*)(w + 1572864);          // 8192*16*4 = 512 KB
    float*  H1n = (float*)(w + 2097152);          // 512 KB
    float*  H2i = (float*)(w + 2621440);          // 8192*32*4 = 1 MB
    float*  H2n = (float*)(w + 3670016);          // 1 MB
    float*  E    = (float*)(w + 4718592);         // 8192*3*4
    float*  part = (float*)(w + 4816896);         // 32*195*4

    // layer 1: Y = X @ W1 (packed bf16), H1 = relu(adj @ Y + b1)
    pack_kernel<<<(NN * 16 + 255) / 256, 256, 0, stream>>>(x_int, W1_int, Y1i, 11, 16);
    pack_kernel<<<(NN * 16 + 255) / 256, 256, 0, stream>>>(x_nh,  W1_nh,  Y1n, 11, 16);
    gemm_agg<1><<<NN / 16, 256, 0, stream>>>(adj_int, Y1i, b1_int, H1i);
    gemm_agg<1><<<NN / 16, 256, 0, stream>>>(adj_nh,  Y1n, b1_nh,  H1n);

    // layer 2: Y2 = H1 @ W2 (packed bf16), H2 = relu(adj @ Y2 + b2)
    pack_kernel<<<(NN * 32 + 255) / 256, 256, 0, stream>>>(H1i, W2_int, Y2i, 16, 32);
    pack_kernel<<<(NN * 32 + 255) / 256, 256, 0, stream>>>(H1n, W2_nh,  Y2n, 16, 32);
    gemm_agg<2><<<NN / 16, 256, 0, stream>>>(adj_int, Y2i, b2_int, H2i);
    gemm_agg<2><<<NN / 16, 256, 0, stream>>>(adj_nh,  Y2n, b2_nh,  H2n);

    // attention pooling + dense head
    escore_kernel<<<NN / 256, 256, 0, stream>>>(H2i, H2n, W_att, E);
    pool_partial<<<32, 256, 0, stream>>>(E, H2i, H2n, part);
    final_kernel<<<1, 256, 0, stream>>>(part, W_dense, b_dense, out);
}
